// Elephant_8160437862822
// MI455X (gfx1250) — matrix operand order repaired and checked by reading, not compiled, not run on hardware
//
#include <hip/hip_runtime.h>

// ---------------------------------------------------------------------------
// CDNA5 (gfx1250) 50-branch gated CNN.
//   prep    -> im2col A-matrix (bf16, K and M padded) + bf16 weight matrix
//   gemm    -> bf16 WMMA GEMM, async global->LDS double-buffered staging
//              (global_load_async_to_lds_b128 + s_wait_asynccnt), fused
//              bias+ReLU+bf16 y-store + deterministic activity partials
//   softmax -> tiny per-batch kernel (fixed-order reduction)
//   gate+pool -> bandwidth-bound pass over bf16 y
//   fc1     -> bf16 WMMA GEMM (fused bias+ReLU), fc2 -> tiny VALU kernel
// ---------------------------------------------------------------------------

typedef __attribute__((ext_vector_type(16))) __bf16 v16bf;
typedef __attribute__((ext_vector_type(8)))  float  v8f;

union Frag { unsigned int u[8]; v16bf v; };

__device__ __forceinline__ unsigned short f2bf(float f) {
  unsigned int u = __float_as_uint(f);
  u += 0x7FFFu + ((u >> 16) & 1u);          // round-to-nearest-even
  return (unsigned short)(u >> 16);
}
__device__ __forceinline__ float bf2f(unsigned short h) {
  return __uint_as_float(((unsigned int)h) << 16);
}
// A-matrix (16x32, 16-bit): lane<16 row M=lane with contiguous K 0..15;
// lanes 16-31 hold K 16..31 (same order as B).
__device__ __forceinline__ int kbaseA(int j, int half) {
  return 2 * j + (half ? 16 : 0);
}
// B-matrix (32x16, 16-bit): lane<16 col N=lane with contiguous K 0..15,
// lanes 16-31 hold K 16..31 (same order as A).
__device__ __forceinline__ int kbaseB(int j, int half) {
  return 2 * j + (half ? 16 : 0);
}

#define LDS_STRIDE 40            // halfwords; 80B rows: aligned 16B chunks, no bank clash

// ---------------------------------------------------------------------------
// im2col (3x3 VALID) to padded bf16 A-matrix: A[b][r<RPI][k<Kpad], zero pads.
// grid = (ceil(RPI*Kpad/256), 1, B)
// ---------------------------------------------------------------------------
__global__ void im2col_bf16(const float* __restrict__ x, unsigned short* __restrict__ A,
                            int Cin, int H, int W, int Ho, int Wo,
                            int Kdim, int Kpad, int RPI)
{
  const int b = blockIdx.z;
  const int e = blockIdx.x * blockDim.x + threadIdx.x;
  if (e >= RPI * Kpad) return;
  const int r = e / Kpad;
  const int k = e - r * Kpad;
  const int Pimg = Ho * Wo;
  float v = 0.f;
  if (r < Pimg && k < Kdim) {
    int ci  = k / 9;
    int rem = k - ci * 9;
    int kh  = rem / 3;
    int kw  = rem - kh * 3;
    int oh  = r / Wo;
    int ow  = r - oh * Wo;
    v = x[((size_t)(b * Cin + ci) * H + (oh + kh)) * W + (ow + kw)];
  }
  A[(size_t)b * RPI * Kpad + e] = f2bf(v);
}

// Weights [N, Kdim] fp32 -> [N, Kpad] bf16 (zero pad)
__global__ void wcvt_bf16(const float* __restrict__ w, unsigned short* __restrict__ o,
                          int N, int Kdim, int Kpad)
{
  const int e = blockIdx.x * blockDim.x + threadIdx.x;
  if (e >= N * Kpad) return;
  const int n = e / Kpad;
  const int k = e - n * Kpad;
  o[e] = (k < Kdim) ? f2bf(w[(size_t)n * Kdim + k]) : (unsigned short)0;
}

// ---------------------------------------------------------------------------
// bf16 GEMM with async double-buffered LDS staging.
// A: [B, RPI, Kpad] bf16 (per-image padded rows), Bw: [Nc, Kpad] bf16.
// block = 128 (4 waves as 2Mx2N); tile = 32(M) x 64(N); 2 WMMA/wave/K-step.
// grid = (Nc/64, RPI/32, B)
// ---------------------------------------------------------------------------
__global__ void __launch_bounds__(128)
gemm_gated_wmma(const unsigned short* __restrict__ A,
                const unsigned short* __restrict__ Bw,
                const float* __restrict__ bias,
                unsigned short* __restrict__ y,     // bf16 [B, Nc, Pimg]
                float* __restrict__ part,
                int RPI, int Kpad, int Cout, int Nc, int Pimg)
{
  const int b  = blockIdx.z;
  const int mt = blockIdx.y;
  const int nt = blockIdx.x;
  const int t    = threadIdx.x;
  const int wave = t >> 5;
  const int lane = t & 31;
  const int half = lane >> 4;
  const int lh   = lane & 15;
  const int wm = wave >> 1;          // M half (0/1)
  const int wn = wave & 1;           // N half (0/1)
  const int n0 = nt * 64;

  __shared__ unsigned short Ab[2][32 * LDS_STRIDE];
  __shared__ unsigned short Bb[2][64 * LDS_STRIDE];

  // async staging addressing: each thread owns one 16B chunk of the 32-row A
  // tile and two 16B chunks of the 64-row B tile.
  const int r4 = t >> 2;             // 0..31
  const int c4 = t & 3;              // 16B chunk within a 64B (32 bf16) row
  const unsigned aOff0 = (unsigned)((((size_t)b * RPI + (size_t)mt * 32 + r4) * Kpad + c4 * 8) * 2);
  const unsigned bOff0 = (unsigned)(((size_t)(n0 + r4)      * Kpad + c4 * 8) * 2);
  const unsigned bOff1 = (unsigned)(((size_t)(n0 + 32 + r4) * Kpad + c4 * 8) * 2);
  const unsigned long long aBase = (unsigned long long)(size_t)A;
  const unsigned long long bBase = (unsigned long long)(size_t)Bw;

  auto issue = [&](int buf, unsigned kbyte) {
    unsigned la  = (unsigned)(size_t)&Ab[buf][r4 * LDS_STRIDE + c4 * 8];
    unsigned lb0 = (unsigned)(size_t)&Bb[buf][r4 * LDS_STRIDE + c4 * 8];
    unsigned lb1 = (unsigned)(size_t)&Bb[buf][(32 + r4) * LDS_STRIDE + c4 * 8];
    asm volatile("global_load_async_to_lds_b128 %0, %1, %2"
                 :: "v"(la),  "v"(aOff0 + kbyte), "s"(aBase) : "memory");
    asm volatile("global_load_async_to_lds_b128 %0, %1, %2"
                 :: "v"(lb0), "v"(bOff0 + kbyte), "s"(bBase) : "memory");
    asm volatile("global_load_async_to_lds_b128 %0, %1, %2"
                 :: "v"(lb1), "v"(bOff1 + kbyte), "s"(bBase) : "memory");
  };

  v8f acc0 = {};
  v8f acc1 = {};
  const int nsteps = Kpad >> 5;

  issue(0, 0u);
  for (int s = 0; s < nsteps; ++s) {
    const int cur = s & 1;
    if (s + 1 < nsteps) {
      issue(1 - cur, (unsigned)((s + 1) * 64));
      asm volatile("s_wait_asynccnt 3" ::: "memory");   // current buffer landed
    } else {
      asm volatile("s_wait_asynccnt 0" ::: "memory");
    }
    __syncthreads();                                    // all waves' data visible

    Frag fa, fb0, fb1;
    const int arow = (wm * 16 + lh) * LDS_STRIDE;
    const int brow0 = (wn * 32 + lh) * LDS_STRIDE;
    const int brow1 = (wn * 32 + 16 + lh) * LDS_STRIDE;
    #pragma unroll
    for (int j = 0; j < 8; ++j) {
      fa.u[j]  = *(const unsigned int*)&Ab[cur][arow  + kbaseA(j, half)];
      fb0.u[j] = *(const unsigned int*)&Bb[cur][brow0 + kbaseB(j, half)];
      fb1.u[j] = *(const unsigned int*)&Bb[cur][brow1 + kbaseB(j, half)];
    }
    acc0 = __builtin_amdgcn_wmma_f32_16x16x32_bf16(
        false, fa.v, false, fb0.v, (short)0, acc0, false, false);
    acc1 = __builtin_amdgcn_wmma_f32_16x16x32_bf16(
        false, fa.v, false, fb1.v, (short)0, acc1, false, false);
    __syncthreads();                                    // done reading `cur`
  }

  // epilogue: bias + ReLU + bf16 store + activity partial
  const int nA = n0 + wn * 32 + lh;      // acc0 column
  const int nB = nA + 16;                // acc1 column
  const float bv0 = bias[nA];
  const float bv1 = bias[nB];
  const int pBase = mt * 32 + wm * 16 + (half ? 8 : 0);
  float asum = 0.f;
  #pragma unroll
  for (int i = 0; i < 8; ++i) {
    const int p = pBase + i;
    float v0 = fmaxf(acc0[i] + bv0, 0.f);
    float v1 = fmaxf(acc1[i] + bv1, 0.f);
    if (p < Pimg) {
      y[((size_t)b * Nc + nA) * Pimg + p] = f2bf(v0);
      y[((size_t)b * Nc + nB) * Pimg + p] = f2bf(v1);
      asum += v0 + v1;
    }
  }
  #pragma unroll
  for (int off = 16; off > 0; off >>= 1)
    asum += __shfl_xor(asum, off, 32);
  if (lane == 0) {
    const int nw     = n0 + wn * 32;
    const int branch = nw / Cout;
    const int ws     = Cout >> 5;                      // 1 (Cout=32) or 2 (Cout=64)
    const int wsub   = (nw - branch * Cout) >> 5;
    const int mteff  = mt * 2 + wm;
    part[(((size_t)b * 50 + branch) * (gridDim.y * 2) + mteff) * ws + wsub] = asum;
  }
}

// ---------------------------------------------------------------------------
// probs[b,k] = softmax(mean_k). Deterministic fixed-order partial reduction.
// grid = B, block = 64. Entries per (b,k) = MT*WS.
// ---------------------------------------------------------------------------
__global__ void gate_softmax(const float* __restrict__ part,
                             float* __restrict__ probs,
                             int MT, int WS, float inv_mean_div)
{
  const int b = blockIdx.x;
  const int t = threadIdx.x;
  __shared__ float s[50];
  __shared__ float sinv;
  if (t < 50) {
    float a = 0.f;
    const float* p = part + ((size_t)b * 50 + t) * MT * WS;
    for (int i = 0; i < MT * WS; ++i) a += p[i];
    s[t] = a * inv_mean_div;                 // temp = 1
  }
  __syncthreads();
  if (t == 0) {
    float m = s[0];
    for (int k = 1; k < 50; ++k) m = fmaxf(m, s[k]);
    float su = 0.f;
    for (int k = 0; k < 50; ++k) { s[k] = __expf(s[k] - m); su += s[k]; }
    sinv = 1.f / su;
  }
  __syncthreads();
  if (t < 50) probs[b * 50 + t] = s[t] * sinv;
}

// ---------------------------------------------------------------------------
// Gated sum over branches then 2x2 max-pool (VALID). One pass over bf16 y.
// grid = (ceil(Cout*Hp*Wp/256), 1, B), block = 256.
// ---------------------------------------------------------------------------
__global__ void gate_pool(const unsigned short* __restrict__ y,
                          const float* __restrict__ probs,
                          float* __restrict__ h,
                          int Cout, int Nc, int Ho, int Wo, int Hp, int Wp)
{
  const int b = blockIdx.z;
  __shared__ float ps[50];
  if (threadIdx.x < 50) ps[threadIdx.x] = probs[b * 50 + threadIdx.x];
  __syncthreads();

  const int tot = Cout * Hp * Wp;
  const int idx = blockIdx.x * blockDim.x + threadIdx.x;
  if (idx >= tot) return;
  const int hw  = Hp * Wp;
  const int c   = idx / hw;
  const int r   = idx - c * hw;
  const int oh2 = r / Wp;
  const int ow2 = r - oh2 * Wp;
  const int Pimg = Ho * Wo;
  const size_t kstride = (size_t)Cout * Pimg;

  float best = -3.402823466e38f;
  #pragma unroll
  for (int dy = 0; dy < 2; ++dy) {
    #pragma unroll
    for (int dx = 0; dx < 2; ++dx) {
      const int oh = oh2 * 2 + dy;
      const int ow = ow2 * 2 + dx;
      size_t base = ((size_t)b * Nc + c) * Pimg + oh * Wo + ow;
      float sgate = 0.f;
      for (int k = 0; k < 50; ++k)
        sgate += ps[k] * bf2f(y[base + (size_t)k * kstride]);
      best = fmaxf(best, sgate);
    }
  }
  h[((size_t)b * Cout + c) * hw + r] = best;
}

// ---------------------------------------------------------------------------
// Dense fp32-input GEMM: out = relu?(A[M,K] * Wn[N,K]^T + bias), bf16 WMMA.
// grid = (N/64, M/16), block = 128 (wave w owns 16 N cols).
// ---------------------------------------------------------------------------
__global__ void fc_wmma(const float* __restrict__ A,
                        const float* __restrict__ Wn,
                        const float* __restrict__ bias,
                        float* __restrict__ out,
                        int M, int N, int Kdim, int doRelu)
{
  const int t    = threadIdx.x;
  const int wave = t >> 5;
  const int lane = t & 31;
  const int half = lane >> 4;
  const int lh   = lane & 15;
  const int m0 = blockIdx.y * 16;
  const int n0 = blockIdx.x * 64;

  __shared__ unsigned short Abuf[16 * 32];
  __shared__ unsigned short Bt[64 * 36];

  v8f acc = {};
  for (int k0 = 0; k0 < Kdim; k0 += 32) {
    #pragma unroll
    for (int i = 0; i < 4; ++i) {
      int idx = t + i * 128;
      int r = idx >> 5, kk = idx & 31;
      int kg = k0 + kk, m = m0 + r;
      float v = (m < M && kg < Kdim) ? A[(size_t)m * Kdim + kg] : 0.f;
      Abuf[idx] = f2bf(v);
    }
    #pragma unroll
    for (int i = 0; i < 16; ++i) {
      int idx = t + i * 128;
      int kk = idx & 31, nn = idx >> 5;
      int kg = k0 + kk;
      float v = (kg < Kdim) ? Wn[(size_t)(n0 + nn) * Kdim + kg] : 0.f;
      Bt[nn * 36 + kk] = f2bf(v);
    }
    __syncthreads();
    Frag fa, fb;
    const int arow = lh * 32;
    const int brow = (wave * 16 + lh) * 36;
    #pragma unroll
    for (int j = 0; j < 8; ++j) {
      fa.u[j] = *(const unsigned int*)&Abuf[arow + kbaseA(j, half)];
      fb.u[j] = *(const unsigned int*)&Bt[brow + kbaseB(j, half)];
    }
    acc = __builtin_amdgcn_wmma_f32_16x16x32_bf16(
        false, fa.v, false, fb.v, (short)0, acc, false, false);
    __syncthreads();
  }

  const int n = n0 + wave * 16 + lh;
  const float bv = bias[n];
  #pragma unroll
  for (int i = 0; i < 8; ++i) {
    int m = m0 + i + (half ? 8 : 0);
    if (m < M) {
      float v = acc[i] + bv;
      if (doRelu) v = fmaxf(v, 0.f);
      out[(size_t)m * N + n] = v;
    }
  }
}

// out[b,j] = bfc2[j] + sum_k A[b,k]*Wfc2[j,k]   (M=128, N=10, K=128)
__global__ void fc2_kernel(const float* __restrict__ A,
                           const float* __restrict__ W2,
                           const float* __restrict__ b2,
                           float* __restrict__ out)
{
  int i = blockIdx.x * blockDim.x + threadIdx.x;
  if (i >= 128 * 10) return;
  int b = i / 10, j = i - b * 10;
  float s = b2[j];
  for (int k = 0; k < 128; ++k) s += A[b * 128 + k] * W2[j * 128 + k];
  out[i] = s;
}

// ---------------------------------------------------------------------------
extern "C" void kernel_launch(void* const* d_in, const int* in_sizes, int n_in,
                              void* d_out, int out_size, void* d_ws, size_t ws_size,
                              hipStream_t stream) {
  const float* x    = (const float*)d_in[0];   // [128,1,28,28]
  const float* W1   = (const float*)d_in[1];   // [50,32,1,3,3]  -> [1600,9]
  const float* b1   = (const float*)d_in[2];   // [1600]
  const float* W2   = (const float*)d_in[3];   // [50,64,32,3,3] -> [3200,288]
  const float* b2   = (const float*)d_in[4];   // [3200]
  const float* Wfc1 = (const float*)d_in[5];   // [128,1600]
  const float* bfc1 = (const float*)d_in[6];   // [128]
  const float* Wfc2 = (const float*)d_in[7];   // [10,128]
  const float* bfc2 = (const float*)d_in[8];   // [10]
  float* out = (float*)d_out;                  // [128,10]
  (void)in_sizes; (void)n_in; (void)out_size; (void)ws_size;

  // ---- workspace layout (256-aligned) ----
  char* base = (char*)d_ws;
  size_t off = 0;
  auto alloc = [&](size_t bytes) { char* p = base + off; off += (bytes + 255) & ~(size_t)255; return p; };
  unsigned short* ybuf = (unsigned short*)alloc((size_t)128 * 1600 * 676 * 2); // reused L2
  unsigned short* Abf  = (unsigned short*)alloc((size_t)128 * 128 * 288 * 2);  // max of L1/L2
  unsigned short* Wbf  = (unsigned short*)alloc((size_t)3200 * 288 * 2);       // max of L1/L2
  float* part   = (float*)alloc((size_t)128 * 50 * 44 * 4);                    // max of L1/L2
  float* probs1 = (float*)alloc(128 * 50 * 4);
  float* probs2 = (float*)alloc(128 * 50 * 4);
  float* h1     = (float*)alloc((size_t)128 * 32 * 13 * 13 * 4);
  float* h2     = (float*)alloc((size_t)128 * 64 * 5 * 5 * 4);                 // == [128,1600]
  float* fc1o   = (float*)alloc((size_t)128 * 128 * 4);

  // ---- layer 1: Cin=1 28x28 -> 26x26, Nc=1600, Kdim=9 (Kpad=32, RPI=704) ----
  {
    im2col_bf16<<<dim3((704 * 32 + 255) / 256, 1, 128), 256, 0, stream>>>(
        x, Abf, 1, 28, 28, 26, 26, 9, 32, 704);
    wcvt_bf16<<<(1600 * 32 + 255) / 256, 256, 0, stream>>>(W1, Wbf, 1600, 9, 32);
    gemm_gated_wmma<<<dim3(1600 / 64, 704 / 32, 128), 128, 0, stream>>>(
        Abf, Wbf, b1, ybuf, part, 704, 32, 32, 1600, 676);
    gate_softmax<<<128, 64, 0, stream>>>(part, probs1, 44, 1, 1.f / (32.f * 676.f));
    gate_pool<<<dim3((32 * 13 * 13 + 255) / 256, 1, 128), 256, 0, stream>>>(
        ybuf, probs1, h1, 32, 1600, 26, 26, 13, 13);
  }

  // ---- layer 2: Cin=32 13x13 -> 11x11, Nc=3200, Kdim=288 (Kpad=288, RPI=128) ----
  {
    im2col_bf16<<<dim3((128 * 288 + 255) / 256, 1, 128), 256, 0, stream>>>(
        h1, Abf, 32, 13, 13, 11, 11, 288, 288, 128);
    wcvt_bf16<<<(3200 * 288 + 255) / 256, 256, 0, stream>>>(W2, Wbf, 3200, 288, 288);
    gemm_gated_wmma<<<dim3(3200 / 64, 128 / 32, 128), 128, 0, stream>>>(
        Abf, Wbf, b2, ybuf, part, 128, 288, 64, 3200, 121);
    gate_softmax<<<128, 64, 0, stream>>>(part, probs2, 8, 2, 1.f / (64.f * 121.f));
    gate_pool<<<dim3((64 * 5 * 5 + 255) / 256, 1, 128), 256, 0, stream>>>(
        ybuf, probs2, h2, 64, 3200, 11, 11, 5, 5);
  }

  // ---- fc1: [128,1600] x [128,1600]^T + bias, ReLU ----
  fc_wmma<<<dim3(128 / 64, 128 / 16), 128, 0, stream>>>(h2, Wfc1, bfc1, fc1o, 128, 128, 1600, 1);

  // ---- fc2 ----
  fc2_kernel<<<(1280 + 255) / 256, 256, 0, stream>>>(fc1o, Wfc2, bfc2, out);
}
